// LSTMAttentionDecoder_3882650435839
// MI455X (gfx1250) — matrix operand order repaired and checked by reading, not compiled, not run on hardware
//
#include <hip/hip_runtime.h>
#include <hip/hip_bf16.h>

// ---------------- problem constants ----------------
#define TY   64
#define TX   512
#define BB   32
#define II   512
#define HH   512
#define CTXD 1024
#define G4   2048   // 4*HH

typedef __attribute__((ext_vector_type(16))) __bf16       v16bf;
typedef __attribute__((ext_vector_type(8)))  float        v8f;
typedef __attribute__((ext_vector_type(8)))  unsigned int v8u;
typedef __attribute__((ext_vector_type(4)))  unsigned int v4u;

#if __has_builtin(__builtin_amdgcn_tanhf)
#define fast_tanh(x) __builtin_amdgcn_tanhf(x)
#else
#define fast_tanh(x) tanhf(x)
#endif

// ---- f32 -> bf16 (RNE) helpers ----
__device__ __forceinline__ unsigned short f2bf(float x) {
  unsigned int u = __float_as_uint(x);
  u = (u + 0x7FFFu + ((u >> 16) & 1u)) >> 16;
  return (unsigned short)u;
}
__device__ __forceinline__ unsigned int bf16pair(float lo, float hi) {
  return (unsigned int)f2bf(lo) | ((unsigned int)f2bf(hi) << 16);
}

// A fragment: 16x32 bf16 (MxK) from pre-converted bf16 row-major A.
// Lane m=L&15 needs 16 contiguous K starting at k0+16*half
// -> one 32-byte block.
__device__ __forceinline__ v16bf load_a_bf16(const unsigned short* A, int lda, int m0, int k0, int lane) {
  int m = m0 + (lane & 15);
  const unsigned short* p = A + (size_t)m * lda + k0 + ((lane >> 4) << 4);
  return __builtin_bit_cast(v16bf, *(const v8u*)p);
}

// B fragment: 32x16 bf16 (KxN) where B[k,n] = W[n,k], W bf16 row-major [N,K].
// Lane n=L&15: 16 contiguous K starting at k0+16*half -> one 32-byte block.
__device__ __forceinline__ v16bf load_bT_bf16(const unsigned short* W, int ldk, int n0, int k0, int lane) {
  int n  = n0 + (lane & 15);
  int kb = k0 + ((lane >> 4) << 4);
  const v8u* p = (const v8u*)(W + (size_t)n * ldk + kb);
  return __builtin_bit_cast(v16bf, *p);
}

__device__ __forceinline__ v8f wmma_bf16(v16bf a, v16bf b, v8f c) {
  return __builtin_amdgcn_wmma_f32_16x16x32_bf16(false, a, false, b, (short)0, c, false, false);
}

__device__ __forceinline__ float sigf(float x) { return 1.0f / (1.0f + __expf(-x)); }

// ---------------- one-time kernels ----------------
__global__ void k_f32_to_bf16(const float* __restrict__ src, unsigned short* __restrict__ dst, int n) {
  int i = blockIdx.x * blockDim.x + threadIdx.x;
  if (i < n) dst[i] = f2bf(src[i]);
}

// vectorized converter: 4 floats -> 2 packed uints per thread (n must be multiple of 4)
__global__ void k_cvt4(const float* __restrict__ src, unsigned int* __restrict__ dst, int n4) {
  int i = blockIdx.x * blockDim.x + threadIdx.x;
  if (i < n4) {
    float4 v = ((const float4*)src)[i];
    dst[2 * i]     = bf16pair(v.x, v.y);
    dst[2 * i + 1] = bf16pair(v.z, v.w);
  }
}

// pctx[m, d] = sum_c context[m, c] * Wc_att[d, c] + b_att[d]; 4 N-tiles per wave
__global__ void k_pctx(const unsigned short* __restrict__ ctxb, const unsigned short* __restrict__ Wc,
                       const float* __restrict__ b_att, float* __restrict__ pctx) {
  int w    = blockIdx.x * (blockDim.x >> 5) + (threadIdx.x >> 5);
  int lane = threadIdx.x & 31;
  int g = w & 15;                 // 16 groups of 4 N-tiles
  int mtile = w >> 4;             // 1024 m-tiles
  int m0 = mtile << 4, n0 = g << 6;
  v8f a0 = {}, a1 = {}, a2 = {}, a3 = {};
#pragma unroll 2
  for (int k0 = 0; k0 < CTXD; k0 += 32) {
    v16bf a  = load_a_bf16(ctxb, CTXD, m0, k0, lane);
    v16bf b0 = load_bT_bf16(Wc, CTXD, n0 +  0, k0, lane);
    v16bf b1 = load_bT_bf16(Wc, CTXD, n0 + 16, k0, lane);
    v16bf b2 = load_bT_bf16(Wc, CTXD, n0 + 32, k0, lane);
    v16bf b3 = load_bT_bf16(Wc, CTXD, n0 + 48, k0, lane);
    a0 = wmma_bf16(a, b0, a0);
    a1 = wmma_bf16(a, b1, a1);
    a2 = wmma_bf16(a, b2, a2);
    a3 = wmma_bf16(a, b3, a3);
  }
  int half = lane >> 4, nl = lane & 15;
#pragma unroll
  for (int j = 0; j < 8; ++j) {
    int m = m0 + j + (half << 3);
    size_t row = (size_t)m * CTXD;
    int n = n0 + nl;
    pctx[row + n]      = a0[j] + b_att[n];
    pctx[row + n + 16] = a1[j] + b_att[n + 16];
    pctx[row + n + 32] = a2[j] + b_att[n + 32];
    pctx[row + n + 48] = a3[j] + b_att[n + 48];
  }
}

// ---------------- per-step kernels ----------------
// Fused LSTM cell: 64 waves, each owns (m-tile, n0 in H) and all 4 gates.
__global__ void k_lstm(const unsigned short* __restrict__ xb, const unsigned short* __restrict__ hb,
                       const float* __restrict__ h_prev, const float* __restrict__ c_prev,
                       const unsigned short* __restrict__ Wih, const unsigned short* __restrict__ Whh,
                       const float* __restrict__ b_ih, const float* __restrict__ b_hh,
                       const float* __restrict__ ym,
                       float* __restrict__ h1, float* __restrict__ c1,
                       unsigned short* __restrict__ h1b, unsigned short* __restrict__ c1b) {
  int w    = blockIdx.x * (blockDim.x >> 5) + (threadIdx.x >> 5);
  int lane = threadIdx.x & 31;
  int mtile = w >> 5, ntile = w & 31;
  int m0 = mtile << 4, n0 = ntile << 4;
  v8f a0 = {}, a1 = {}, a2 = {}, a3 = {};
#pragma unroll 2
  for (int k0 = 0; k0 < II; k0 += 32) {
    v16bf a  = load_a_bf16(xb, II, m0, k0, lane);
    v16bf b0 = load_bT_bf16(Wih, II, 0 * HH + n0, k0, lane);
    v16bf b1 = load_bT_bf16(Wih, II, 1 * HH + n0, k0, lane);
    v16bf b2 = load_bT_bf16(Wih, II, 2 * HH + n0, k0, lane);
    v16bf b3 = load_bT_bf16(Wih, II, 3 * HH + n0, k0, lane);
    a0 = wmma_bf16(a, b0, a0);
    a1 = wmma_bf16(a, b1, a1);
    a2 = wmma_bf16(a, b2, a2);
    a3 = wmma_bf16(a, b3, a3);
  }
#pragma unroll 2
  for (int k0 = 0; k0 < HH; k0 += 32) {
    v16bf a  = load_a_bf16(hb, HH, m0, k0, lane);
    v16bf b0 = load_bT_bf16(Whh, HH, 0 * HH + n0, k0, lane);
    v16bf b1 = load_bT_bf16(Whh, HH, 1 * HH + n0, k0, lane);
    v16bf b2 = load_bT_bf16(Whh, HH, 2 * HH + n0, k0, lane);
    v16bf b3 = load_bT_bf16(Whh, HH, 3 * HH + n0, k0, lane);
    a0 = wmma_bf16(a, b0, a0);
    a1 = wmma_bf16(a, b1, a1);
    a2 = wmma_bf16(a, b2, a2);
    a3 = wmma_bf16(a, b3, a3);
  }
  int half = lane >> 4, n = n0 + (lane & 15);
  float bi = b_ih[n] + b_hh[n];
  float bf = b_ih[HH + n] + b_hh[HH + n];
  float bg = b_ih[2 * HH + n] + b_hh[2 * HH + n];
  float bo = b_ih[3 * HH + n] + b_hh[3 * HH + n];
#pragma unroll
  for (int j = 0; j < 8; ++j) {
    int m = m0 + j + (half << 3);          // batch index
    float gi = a0[j] + bi, gf = a1[j] + bf, gg = a2[j] + bg, go = a3[j] + bo;
    float cp = c_prev[m * HH + n], hp = h_prev[m * HH + n];
    float cn = sigf(gf) * cp + sigf(gi) * fast_tanh(gg);
    float hn = sigf(go) * fast_tanh(cn);
    float yv = ym[m];
    hn = yv * hn + (1.0f - yv) * hp;
    cn = yv * cn + (1.0f - yv) * cp;
    h1[m * HH + n] = hn;  c1[m * HH + n] = cn;
    h1b[m * HH + n] = f2bf(hn);  c1b[m * HH + n] = f2bf(cn);
  }
}

// hcomb = [h1, c1] @ W_combT : M=32, N=1024, K=1024; 4 N-tiles per wave (32 waves)
__global__ void k_hcomb(const unsigned short* __restrict__ h1b, const unsigned short* __restrict__ c1b,
                        const unsigned short* __restrict__ Wcomb, float* __restrict__ hcomb) {
  int w    = blockIdx.x * (blockDim.x >> 5) + (threadIdx.x >> 5);
  int lane = threadIdx.x & 31;
  int g = w & 15, mtile = w >> 4;
  int m0 = mtile << 4, n0 = g << 6;
  v8f a0 = {}, a1 = {}, a2 = {}, a3 = {};
#pragma unroll 2
  for (int k0 = 0; k0 < 2 * HH; k0 += 32) {
    const unsigned short* Asrc = (k0 < HH) ? h1b : c1b;
    int kk = (k0 < HH) ? k0 : (k0 - HH);
    v16bf a  = load_a_bf16(Asrc, HH, m0, kk, lane);
    v16bf b0 = load_bT_bf16(Wcomb, 2 * HH, n0 +  0, k0, lane);
    v16bf b1 = load_bT_bf16(Wcomb, 2 * HH, n0 + 16, k0, lane);
    v16bf b2 = load_bT_bf16(Wcomb, 2 * HH, n0 + 32, k0, lane);
    v16bf b3 = load_bT_bf16(Wcomb, 2 * HH, n0 + 48, k0, lane);
    a0 = wmma_bf16(a, b0, a0);
    a1 = wmma_bf16(a, b1, a1);
    a2 = wmma_bf16(a, b2, a2);
    a3 = wmma_bf16(a, b3, a3);
  }
  int half = lane >> 4, nl = lane & 15;
#pragma unroll
  for (int j = 0; j < 8; ++j) {
    int m = m0 + j + (half << 3);
    size_t row = (size_t)m * CTXD;
    int n = n0 + nl;
    hcomb[row + n]      = a0[j];
    hcomb[row + n + 16] = a1[j];
    hcomb[row + n + 32] = a2[j];
    hcomb[row + n + 48] = a3[j];
  }
}

// u[t,b] = x_mask[t,b] * sum_d tanh(pctx[t,b,d] + hcomb[b,d]) * U_att[d]   (wave per (t,b))
__global__ void k_scores(const float* __restrict__ pctx, const float* __restrict__ hcomb,
                         const float* __restrict__ U_att, const float* __restrict__ x_mask,
                         float* __restrict__ u) {
  int w    = blockIdx.x * (blockDim.x >> 5) + (threadIdx.x >> 5);
  int lane = threadIdx.x & 31;
  int t = w >> 5, b = w & 31;            // BB == 32
  const float4* pr = (const float4*)(pctx + (size_t)(t * BB + b) * CTXD);
  const float4* hr = (const float4*)(hcomb + (size_t)b * CTXD);
  const float4* Ua = (const float4*)U_att;
  float acc = 0.0f;
#pragma unroll 2
  for (int d = lane; d < CTXD / 4; d += 32) {
    float4 pv = pr[d], hv = hr[d], uv = Ua[d];
    acc += fast_tanh(pv.x + hv.x) * uv.x + fast_tanh(pv.y + hv.y) * uv.y
         + fast_tanh(pv.z + hv.z) * uv.z + fast_tanh(pv.w + hv.w) * uv.w;
  }
#pragma unroll
  for (int off = 16; off; off >>= 1) acc += __shfl_xor(acc, off, 32);
  if (lane == 0) u[t * BB + b] = acc * x_mask[t * BB + b];
}

// per-batch softmax over Tx (block per b, 512 threads == TX)
__global__ void k_softmax(const float* __restrict__ u, const float* __restrict__ x_mask,
                          float* __restrict__ a_un, float* __restrict__ inv_sum) {
  __shared__ float red[TX];
  int b = blockIdx.x, t = threadIdx.x;
  float v = u[t * BB + b];
  red[t] = v; __syncthreads();
  for (int s = TX / 2; s; s >>= 1) { if (t < s) red[t] = fmaxf(red[t], red[t + s]); __syncthreads(); }
  float mx = red[0]; __syncthreads();
  float e = __expf(v - mx) * x_mask[t * BB + b];
  red[t] = e; __syncthreads();
  for (int s = TX / 2; s; s >>= 1) { if (t < s) red[t] += red[t + s]; __syncthreads(); }
  a_un[t * BB + b] = e;
  if (t == 0) inv_sum[b] = 1.0f / red[0];
}

// atted[b,c] = inv_sum[b] * sum_t a_un[t,b] * context[t,b,c]; wave per (b, 128-col chunk)
__global__ void k_atted(const float* __restrict__ a_un, const float* __restrict__ inv_sum,
                        const float* __restrict__ context,
                        float* __restrict__ atted, unsigned short* __restrict__ atted_bf) {
  int w    = blockIdx.x * (blockDim.x >> 5) + (threadIdx.x >> 5);
  int lane = threadIdx.x & 31;
  int b = w >> 3;                          // 8 chunks per b -> 256 waves
  int c = ((w & 7) << 7) + (lane << 2);
  float ax = 0.f, ay = 0.f, az = 0.f, aw = 0.f;
  const float* cb = context + (size_t)b * CTXD + c;
  for (int t = 0; t < TX; ++t) {
    float av = a_un[t * BB + b];
    float4 cv = *(const float4*)(cb + (size_t)t * BB * CTXD);
    ax += av * cv.x; ay += av * cv.y; az += av * cv.z; aw += av * cv.w;
  }
  float s = inv_sum[b];
  ax *= s; ay *= s; az *= s; aw *= s;
  size_t o = (size_t)b * CTXD + c;
  float4 r; r.x = ax; r.y = ay; r.z = az; r.w = aw;
  *(float4*)(atted + o) = r;
  unsigned int* mb = (unsigned int*)atted_bf;
  mb[o >> 1]       = bf16pair(ax, ay);
  mb[(o >> 1) + 1] = bf16pair(az, aw);
}

// second gated update (gate order i, f, o, c) — fused across 4 gates per wave
__global__ void k_update2(const unsigned short* __restrict__ h1b, const unsigned short* __restrict__ attb,
                          const float* __restrict__ h1, const float* __restrict__ c1,
                          const unsigned short* __restrict__ Uxb, const unsigned short* __restrict__ Wxb,
                          const float* __restrict__ bx, const float* __restrict__ ym,
                          float* __restrict__ h_state, float* __restrict__ c_state,
                          unsigned short* __restrict__ h_state_bf,
                          float* __restrict__ out_h, float* __restrict__ out_c,
                          float* __restrict__ out_s) {
  int w    = blockIdx.x * (blockDim.x >> 5) + (threadIdx.x >> 5);
  int lane = threadIdx.x & 31;
  int mtile = w >> 5, ntile = w & 31;
  int m0 = mtile << 4, n0 = ntile << 4;
  v8f a0 = {}, a1 = {}, a2 = {}, a3 = {};
#pragma unroll 2
  for (int k0 = 0; k0 < HH; k0 += 32) {
    v16bf a  = load_a_bf16(h1b, HH, m0, k0, lane);
    v16bf b0 = load_bT_bf16(Uxb, HH, 0 * HH + n0, k0, lane);
    v16bf b1 = load_bT_bf16(Uxb, HH, 1 * HH + n0, k0, lane);
    v16bf b2 = load_bT_bf16(Uxb, HH, 2 * HH + n0, k0, lane);
    v16bf b3 = load_bT_bf16(Uxb, HH, 3 * HH + n0, k0, lane);
    a0 = wmma_bf16(a, b0, a0);
    a1 = wmma_bf16(a, b1, a1);
    a2 = wmma_bf16(a, b2, a2);
    a3 = wmma_bf16(a, b3, a3);
  }
#pragma unroll 2
  for (int k0 = 0; k0 < CTXD; k0 += 32) {
    v16bf a  = load_a_bf16(attb, CTXD, m0, k0, lane);
    v16bf b0 = load_bT_bf16(Wxb, CTXD, 0 * HH + n0, k0, lane);
    v16bf b1 = load_bT_bf16(Wxb, CTXD, 1 * HH + n0, k0, lane);
    v16bf b2 = load_bT_bf16(Wxb, CTXD, 2 * HH + n0, k0, lane);
    v16bf b3 = load_bT_bf16(Wxb, CTXD, 3 * HH + n0, k0, lane);
    a0 = wmma_bf16(a, b0, a0);
    a1 = wmma_bf16(a, b1, a1);
    a2 = wmma_bf16(a, b2, a2);
    a3 = wmma_bf16(a, b3, a3);
  }
  int half = lane >> 4, n = n0 + (lane & 15);
  float bi = bx[n], bf = bx[HH + n], bo = bx[2 * HH + n], bc = bx[3 * HH + n];
#pragma unroll
  for (int j = 0; j < 8; ++j) {
    int m = m0 + j + (half << 3);
    float xi = a0[j] + bi, xf = a1[j] + bf, xo = a2[j] + bo, xc = a3[j] + bc;
    float c1v = c1[m * HH + n], h1v = h1[m * HH + n];
    float c2 = sigf(xf) * c1v + sigf(xi) * fast_tanh(xc);
    float h2 = sigf(xo) * fast_tanh(c2);
    float yv = ym[m];
    c2 = yv * c2 + (1.0f - yv) * c1v;
    h2 = yv * h2 + (1.0f - yv) * h1v;
    c_state[m * HH + n] = c2;
    h_state[m * HH + n] = h2;
    h_state_bf[m * HH + n] = f2bf(h2);
    out_h[m * HH + n] = h2;
    out_c[m * HH + n] = c2;
    out_s[m * HH + n] = h2;
  }
}

// ---------------- host launcher ----------------
extern "C" void kernel_launch(void* const* d_in, const int* in_sizes, int n_in,
                              void* d_out, int out_size, void* d_ws, size_t ws_size,
                              hipStream_t stream) {
  const float* y_emb   = (const float*)d_in[0];
  const float* context = (const float*)d_in[1];
  const float* h0      = (const float*)d_in[2];
  const float* c0      = (const float*)d_in[3];
  const float* x_mask  = (const float*)d_in[4];
  const float* y_mask  = (const float*)d_in[5];
  const float* W_ih    = (const float*)d_in[6];
  const float* W_hh    = (const float*)d_in[7];
  const float* b_ih    = (const float*)d_in[8];
  const float* b_hh    = (const float*)d_in[9];
  const float* Wx      = (const float*)d_in[10];
  const float* Ux      = (const float*)d_in[11];
  const float* bx      = (const float*)d_in[12];
  const float* Wc_att  = (const float*)d_in[13];
  const float* b_att   = (const float*)d_in[14];
  const float* W_comb  = (const float*)d_in[15];
  const float* U_att   = (const float*)d_in[16];
  (void)in_sizes; (void)n_in; (void)out_size; (void)ws_size;

  float* out      = (float*)d_out;
  float* out_hs   = out;
  float* out_cs   = out + (size_t)TY * BB * HH;
  float* out_ss   = out + 2 * (size_t)TY * BB * HH;
  float* out_atts = out + 3 * (size_t)TY * BB * HH;

  // workspace carve-out (~115 MB): bf16 weights (14MB) + bf16 context (32MB)
  // + pctx (64MB) + bf16 y_emb (2MB) + state
  char* p = (char*)d_ws;
  auto carve = [&](size_t bytes) -> char* {
    char* r = p; p += (bytes + 255) & ~(size_t)255; return r;
  };
  unsigned short* Wih_b   = (unsigned short*)carve((size_t)G4 * II * 2);
  unsigned short* Whh_b   = (unsigned short*)carve((size_t)G4 * HH * 2);
  unsigned short* Wc_b    = (unsigned short*)carve((size_t)CTXD * CTXD * 2);
  unsigned short* Wcomb_b = (unsigned short*)carve((size_t)CTXD * CTXD * 2);
  unsigned short* Ux_b    = (unsigned short*)carve((size_t)G4 * HH * 2);
  unsigned short* Wx_b    = (unsigned short*)carve((size_t)G4 * CTXD * 2);
  unsigned short* yemb_b  = (unsigned short*)carve((size_t)TY * BB * II * 2);
  unsigned short* ctx_b   = (unsigned short*)carve((size_t)TX * BB * CTXD * 2);
  float* pctx    = (float*)carve((size_t)TX * BB * CTXD * 4);
  float* hA      = (float*)carve((size_t)BB * HH * 4);
  float* cA      = (float*)carve((size_t)BB * HH * 4);
  float* hB      = (float*)carve((size_t)BB * HH * 4);
  float* cB      = (float*)carve((size_t)BB * HH * 4);
  unsigned short* hA_b  = (unsigned short*)carve((size_t)BB * HH * 2);
  unsigned short* hB_b  = (unsigned short*)carve((size_t)BB * HH * 2);
  unsigned short* cB_b  = (unsigned short*)carve((size_t)BB * HH * 2);
  unsigned short* att_b = (unsigned short*)carve((size_t)BB * CTXD * 2);
  float* hcomb   = (float*)carve((size_t)BB * CTXD * 4);
  float* uu      = (float*)carve((size_t)TX * BB * 4);
  float* a_un    = (float*)carve((size_t)TX * BB * 4);
  float* inv_sum = (float*)carve((size_t)BB * 4);

  // one-time conversions (all sizes multiple of 4 -> vectorized)
  auto cvt = [&](const float* s, unsigned short* d, int n) {
    int n4 = n >> 2;
    k_cvt4<<<(n4 + 255) / 256, 256, 0, stream>>>(s, (unsigned int*)d, n4);
  };
  cvt(W_ih, Wih_b, G4 * II);
  cvt(W_hh, Whh_b, G4 * HH);
  cvt(Wc_att, Wc_b, CTXD * CTXD);
  cvt(W_comb, Wcomb_b, CTXD * CTXD);
  cvt(Ux, Ux_b, G4 * HH);
  cvt(Wx, Wx_b, G4 * CTXD);
  cvt(y_emb, yemb_b, TY * BB * II);
  cvt(context, ctx_b, TX * BB * CTXD);
  cvt(h0, hA_b, BB * HH);

  hipMemcpyAsync(hA, h0, (size_t)BB * HH * 4, hipMemcpyDeviceToDevice, stream);
  hipMemcpyAsync(cA, c0, (size_t)BB * HH * 4, hipMemcpyDeviceToDevice, stream);

  // one-time: pctx GEMM (16384 waves, 4 N-tiles each)
  k_pctx<<<2048, 256, 0, stream>>>(ctx_b, Wc_b, b_att, pctx);

  for (int t = 0; t < TY; ++t) {
    const unsigned short* xb = yemb_b + (size_t)t * BB * II;
    const float* ym  = y_mask + (size_t)t * BB;
    float* atted_t = out_atts + (size_t)t * BB * CTXD;

    k_lstm<<<16, 128, 0, stream>>>(xb, hA_b, hA, cA, Wih_b, Whh_b, b_ih, b_hh, ym,
                                   hB, cB, hB_b, cB_b);
    k_hcomb<<<8, 128, 0, stream>>>(hB_b, cB_b, Wcomb_b, hcomb);
    k_scores<<<2048, 256, 0, stream>>>(pctx, hcomb, U_att, x_mask, uu);
    k_softmax<<<BB, TX, 0, stream>>>(uu, x_mask, a_un, inv_sum);
    k_atted<<<32, 256, 0, stream>>>(a_un, inv_sum, context, atted_t, att_b);
    k_update2<<<16, 128, 0, stream>>>(hB_b, att_b, hB, cB, Ux_b, Wx_b, bx, ym,
                                      hA, cA, hA_b,
                                      out_hs + (size_t)t * BB * HH,
                                      out_cs + (size_t)t * BB * HH,
                                      out_ss + (size_t)t * BB * HH);
  }
}
